// CNN_symmetry_55765855371906
// MI455X (gfx1250) — matrix operand order repaired and checked by reading, not compiled, not run on hardware
//
#include <hip/hip_runtime.h>

typedef __attribute__((ext_vector_type(16))) _Float16 v16h;
typedef __attribute__((ext_vector_type(8)))  float    v8f;

#define B_    32768
#define NSQ   100
#define LIN_IN 144
#define CELLS 144
#define K1P   160   // feat K padded 144->160 (5 k-steps of 32)
#define N1P   112   // layer1/2 N padded 100->112 (7 n-tiles)
#define LDO   128   // out1/out2 leading dim (4 k-steps of 32 for next layer)

// ---------------- prep: |W| conv kernels + f16-padded MLP weights ----------------
__global__ void prep_kernel(const float* w_each, const float* w_ne, const float* w_ne2,
                            const float* w_emp, const float* W1, const float* W2,
                            const float* b2,
                            float* kern, _Float16* W1p, _Float16* W2p, float* b2p) {
  int tid = blockIdx.x * blockDim.x + threadIdx.x;
  int nth = gridDim.x * blockDim.x;
  const float* srcs[4] = {w_each, w_ne, w_ne2, w_emp};
  for (int i = tid; i < 4 * 91; i += nth) {
    int which = i / 91, t = i % 91, r = t / 7, d = t % 7 - 3;
    int j = d < 0 ? -d : d;                   // K[r][3+d] = |W[r][|d|]|
    kern[i] = fabsf(srcs[which][r * 4 + j]);
  }
  for (int i = tid; i < N1P * K1P; i += nth) {
    int n = i / K1P, k = i % K1P;
    W1p[i] = (_Float16)((n < NSQ && k < LIN_IN) ? W1[n * LIN_IN + k] : 0.f);
  }
  for (int i = tid; i < N1P * LDO; i += nth) {
    int n = i / LDO, k = i % LDO;
    W2p[i] = (_Float16)((n < NSQ && k < NSQ) ? W2[n * NSQ + k] : 0.f);
  }
  for (int i = tid; i < N1P; i += nth) b2p[i] = (i < NSQ) ? b2[i] : 0.f;
}

// ---------------- conv pipeline: 32 images per block (lane = batch) ----------------
__global__ __launch_bounds__(256) void conv_kernel(const int* __restrict__ dots,
                                                   const float* __restrict__ kern,
                                                   _Float16* __restrict__ feat) {
  extern __shared__ char smem[];
  float* s_kern = (float*)smem;                       // 364 f32  (pad to 1536B)
  int*   s_col  = (int*)(smem + 1536);                // [144][32]
  float* s_emp  = (float*)(smem + 1536 + 18432);      // [144][32]
  float* s_vc   = s_emp + CELLS * 32;                 // [144][32]
  float* s_allv = s_vc + CELLS * 32;                  // [144][32]
  float* s_nec  = s_allv + CELLS * 32;                // [5][144][32]

  const int lane = threadIdx.x;              // batch within group of 32
  const int ty   = threadIdx.y;              // 0..7, owns 18 cells
  const int b    = blockIdx.x * 32 + lane;
  const int tlin = ty * 32 + lane;

  for (int i = tlin; i < 364; i += 256) s_kern[i] = kern[i];
  // dots layout (V,H,B): element (v,h,b) at (v*12+h)*B + b ; rows 12,13 dropped
  for (int i = 0; i < 18; ++i) {
    int cell = ty * 18 + i;
    s_col[cell * 32 + lane] = dots[(size_t)cell * B_ + b];   // coalesced over lanes
  }
  __syncthreads();

  const float* Keach = s_kern;
  const float* Kne   = s_kern + 91;
  const float* Kne2  = s_kern + 182;
  const float* Kemp  = s_kern + 273;

  // ---- pass 1: one sweep over taps reading neighbor color ----
  for (int i = 0; i < 18; ++i) {
    int cell = ty * 18 + i;
    int y = cell / 12, x = cell % 12;
    int idx = cell * 32 + lane;
    int c0 = s_col[idx];
    float emp = 0.f, ne_tot = 0.f, each_own = 0.f;
    float ne0 = 0.f, ne1 = 0.f, ne2 = 0.f, ne3 = 0.f, ne4 = 0.f;
    for (int dr = -6; dr <= 6; ++dr) {
      int yy = y + dr;
      if (yy < 0 || yy >= 12) continue;
      int t0 = (dr + 6) * 7;
#pragma unroll
      for (int dc = -3; dc <= 3; ++dc) {
        int xx = x + dc;
        if (xx < 0 || xx >= 12) continue;   // zero padding == skip
        int cn = s_col[(yy * 12 + xx) * 32 + lane];
        int t = t0 + dc + 3;
        float kne = Kne[t];
        if (cn == 0) {
          emp += Kemp[t];
        } else {
          ne_tot += kne;
          if (cn == c0) each_own += Keach[t];
          if      (cn == 1) ne0 += kne;
          else if (cn == 2) ne1 += kne;
          else if (cn == 3) ne2 += kne;
          else if (cn == 4) ne3 += kne;
          else              ne4 += kne;
        }
      }
    }
    s_emp[idx] = emp;
    s_nec[0 * (CELLS * 32) + idx] = ne_tot - ne0;
    s_nec[1 * (CELLS * 32) + idx] = ne_tot - ne1;
    s_nec[2 * (CELLS * 32) + idx] = ne_tot - ne2;
    s_nec[3 * (CELLS * 32) + idx] = ne_tot - ne3;
    s_nec[4 * (CELLS * 32) + idx] = ne_tot - ne4;
    float vc = 0.f;
    if (c0 > 0) {
      float ne_own = (c0 == 1) ? ne0 : (c0 == 2) ? ne1 : (c0 == 3) ? ne2 : (c0 == 4) ? ne3 : ne4;
      vc = each_own + emp - (ne_tot - ne_own);
    }
    s_vc[idx] = vc;
    s_allv[idx] = 1.f / (1.f + __expf(-vc));   // empty cell -> sigmoid(0)=0.5 (matches ref)
  }
  __syncthreads();

  // ---- 2 refinement iterations: nec2 = conv(nec_conv[c0] * all_v, K_ne2) ----
  for (int it = 0; it < 2; ++it) {
    for (int i = 0; i < 18; ++i) {
      int cell = ty * 18 + i;
      int idx = cell * 32 + lane;
      int c0 = s_col[idx];
      float vc = 0.f;
      if (c0 > 0) {
        int y = cell / 12, x = cell % 12;
        const float* necc = s_nec + (c0 - 1) * (CELLS * 32);
        float acc = 0.f;
        for (int dr = -6; dr <= 6; ++dr) {
          int yy = y + dr;
          if (yy < 0 || yy >= 12) continue;
          int t0 = (dr + 6) * 7;
#pragma unroll
          for (int dc = -3; dc <= 3; ++dc) {
            int xx = x + dc;
            if (xx < 0 || xx >= 12) continue;
            int nidx = (yy * 12 + xx) * 32 + lane;
            acc += Kne2[t0 + dc + 3] * necc[nidx] * s_allv[nidx];
          }
        }
        vc = s_vc[idx] + s_emp[idx] + acc;     // v accumulates across iterations
      }
      s_vc[idx] = vc;                          // only owner touches s_vc: safe pre-barrier
    }
    __syncthreads();                            // all old all_v reads done
    for (int i = 0; i < 18; ++i) {
      int idx = (ty * 18 + i) * 32 + lane;
      s_allv[idx] = 2.f / (1.f + __expf(-s_vc[idx])) - 1.f;   // sigmoid*2-1
    }
    __syncthreads();
  }

  // ---- feat = final all_v, f16, row-major (B, 160) zero-padded ----
  for (int i = 0; i < 18; ++i) {
    int cell = ty * 18 + i;
    feat[(size_t)b * K1P + cell] = (_Float16)s_allv[cell * 32 + lane];
  }
  feat[(size_t)b * K1P + 144 + ty * 2]     = (_Float16)0.f;   // pad K 144..159
  feat[(size_t)b * K1P + 144 + ty * 2 + 1] = (_Float16)0.f;
}

// ---------------- WMMA GEMM layer: Out = lrelu(A @ Bw^T + bias) ----------------
// A: (B, lda) f16 row-major. Bw: (>=n_tiles*16, ldb) f16 row-major holding W[n][k].
// One 16x16 tile per wave; KSTEPS unrolled v_wmma_f32_16x16x32_f16.
template <int KSTEPS>
__global__ __launch_bounds__(256) void mlp_wmma(const _Float16* __restrict__ A, int lda,
                                                const _Float16* __restrict__ Bw, int ldb,
                                                const float* __restrict__ bias,
                                                _Float16* __restrict__ Out, int ldo,
                                                int n_tiles) {
  const int lane = threadIdx.x;
  const int tile = blockIdx.x * 8 + threadIdx.y;
  const int m0 = (tile / n_tiles) * 16;
  const int n0 = (tile % n_tiles) * 16;
  const int lm = lane & 15;
  const int hi = lane >> 4;

  v8f acc = {};
  const _Float16* arow = A  + (size_t)(m0 + lm) * lda;   // A lane holds row M = lane%16
  const _Float16* brow = Bw + (size_t)(n0 + lm) * ldb;   // B lane holds col N = lane%16
#pragma unroll
  for (int ks = 0; ks < KSTEPS; ++ks) {
    int k0 = ks * 32;
    v16h a, bf;
#pragma unroll
    for (int j = 0; j < 8; ++j) {
      // A 16x32 f16 layout: VGPR j -> K = 2j (j<4) / 16+2(j-4) (j>=4), +8 for lanes 16..31
      int ka = k0 + 2 * j + ((j >= 4) ? 8 : 0) + hi * 8;
      a[2 * j]     = arow[ka];
      a[2 * j + 1] = arow[ka + 1];
      // B 32x16 f16 layout: same K mapping as A: VGPR j -> K = 2j (j<4) / 16+2(j-4) (j>=4), +8 for lanes 16..31
      int kb = k0 + 2 * j + ((j >= 4) ? 8 : 0) + hi * 8;
      bf[2 * j]     = brow[kb];
      bf[2 * j + 1] = brow[kb + 1];
    }
    acc = __builtin_amdgcn_wmma_f32_16x16x32_f16(false, a, false, bf,
                                                 (short)0, acc, false, false);
  }
  float bn = bias ? bias[n0 + lm] : 0.f;
#pragma unroll
  for (int r = 0; r < 8; ++r) {                 // C/D: VGPR r -> row r + 8*hi, col = lane%16
    float v = acc[r] + bn;
    v = v > 0.f ? v : 0.2f * v;                 // leaky relu 0.2
    Out[(size_t)(m0 + r + 8 * hi) * ldo + n0 + lm] = (_Float16)v;
  }
  if (n0 == (n_tiles - 1) * 16) {               // zero-pad cols n_tiles*16 .. ldo-1
    for (int c = n_tiles * 16 + lm; c < ldo; c += 16)
#pragma unroll
      for (int r = 0; r < 8; ++r)
        Out[(size_t)(m0 + r + 8 * hi) * ldo + c] = (_Float16)0.f;
  }
}

// ---------------- final (B,100) . W3 + b3 ----------------
__global__ void final_kernel(const _Float16* __restrict__ out2,
                             const float* __restrict__ W3, const float* __restrict__ b3,
                             float* __restrict__ out) {
  int b = blockIdx.x * blockDim.x + threadIdx.x;
  if (b >= B_) return;
  const _Float16* row = out2 + (size_t)b * LDO;
  float acc = 0.f;
#pragma unroll 4
  for (int k = 0; k < NSQ; ++k) acc += (float)row[k] * W3[k];
  out[b] = acc + b3[0];
}

extern "C" void kernel_launch(void* const* d_in, const int* in_sizes, int n_in,
                              void* d_out, int out_size, void* d_ws, size_t ws_size,
                              hipStream_t stream) {
  const int*   dots   = (const int*)d_in[0];
  const float* w_each = (const float*)d_in[1];
  const float* w_ne   = (const float*)d_in[2];
  const float* w_ne2  = (const float*)d_in[3];
  const float* w_emp  = (const float*)d_in[4];
  const float* W1     = (const float*)d_in[5];
  const float* W2     = (const float*)d_in[6];
  const float* b2     = (const float*)d_in[7];
  const float* W3     = (const float*)d_in[8];
  const float* b3     = (const float*)d_in[9];

  char* ws = (char*)d_ws;
  float*    kern = (float*)(ws + 0);              // 364 f32
  _Float16* W1p  = (_Float16*)(ws + 2048);        // 112*160*2 = 35840
  _Float16* W2p  = (_Float16*)(ws + 37888);       // 112*128*2 = 28672
  float*    b2p  = (float*)(ws + 66560);          // 112 f32
  _Float16* feat = (_Float16*)(ws + 67072);       // 32768*160*2 = 10485760
  _Float16* out1 = (_Float16*)(ws + 10552832);    // 32768*128*2 = 8388608
  _Float16* out2 = (_Float16*)(ws + 18941440);    // 32768*128*2 = 8388608

  prep_kernel<<<64, 256, 0, stream>>>(w_each, w_ne, w_ne2, w_emp, W1, W2, b2,
                                      kern, W1p, W2p, b2p);

  // 167,424 B dynamic LDS (CDNA5 WGP: 320 KB)
  size_t smem = 1536 + (size_t)18432 * 4 + (size_t)5 * 18432;
  conv_kernel<<<B_ / 32, dim3(32, 8), smem, stream>>>(dots, kern, feat);

  int tiles = (B_ / 16) * 7;                      // 14336 tiles, 8 waves/block
  mlp_wmma<5><<<tiles / 8, dim3(32, 8), 0, stream>>>(feat, K1P, W1p, K1P, nullptr,
                                                     out1, LDO, 7);
  mlp_wmma<4><<<tiles / 8, dim3(32, 8), 0, stream>>>(out1, LDO, W2p, LDO, b2p,
                                                     out2, LDO, 7);
  final_kernel<<<B_ / 256, 256, 0, stream>>>(out2, W3, b3, (float*)d_out);
}
